// MP_Layer_40922448396321
// MI455X (gfx1250) — matrix operand order repaired and checked by reading, not compiled, not run on hardware
//
#include <hip/hip_runtime.h>
#include <hip/hip_bf16.h>
#include <math.h>

// Problem constants (B=32, N=96, D=32, F=16)
#define PB 32
#define PN 96
#define PD 32
#define PF 16

typedef __attribute__((ext_vector_type(16))) _Float16 v16h;
typedef __attribute__((ext_vector_type(8)))  float    v8f;

// Single-instruction ReLU: v_med3(x, 0, +inf), no canonicalize.
__device__ __forceinline__ float relu1(float x) {
    return __builtin_amdgcn_fmed3f(x, 0.0f, __builtin_inff());
}

// ---------------------------------------------------------------------------
// CDNA5 async copy: memory -> LDS, 16B per lane, tracked by ASYNCcnt.
// VDST operand is the wave-relative LDS byte offset; the low 32 bits of a
// flat LDS address are exactly that offset (ISA §10.2).
// ---------------------------------------------------------------------------
__device__ __forceinline__ void async_copy_b128(void* lds_ptr, const void* gptr) {
    unsigned lds_off = (unsigned)(uintptr_t)lds_ptr;
    asm volatile("global_load_async_to_lds_b128 %0, %1, off"
                 :: "v"(lds_off), "v"(gptr) : "memory");
}
__device__ __forceinline__ void wait_asynccnt0() {
    asm volatile("s_wait_asynccnt 0x0" ::: "memory");
}

// ---------------------------------------------------------------------------
// Kernel 0: pre-swizzle W_agg [16, 1024] f32 into WMMA B-fragment layout, f16.
// Layout: bprep[(ct*32 + lane)*16 + kk]
//   column c = ct*16+(lane&15); kk < 8 : K = 8*(lane>>4) + kk (rows 0..15 of W_agg)
//   kk >= 8 : zeros (K = 16..31 padding of the 16x16x32 WMMA)
// ---------------------------------------------------------------------------
__global__ __launch_bounds__(256) void prep_wagg_kernel(const float* __restrict__ W_agg,
                                                        _Float16* __restrict__ bprep) {
    int t = blockIdx.x * blockDim.x + threadIdx.x;   // 0 .. 2047
    if (t >= 64 * 32) return;
    int ct = t >> 5, lane = t & 31;
    _Float16* dst = bprep + (size_t)t * 16;
    {
        int c = ct * 16 + (lane & 15);
        #pragma unroll
        for (int kk = 0; kk < 16; ++kk) dst[kk] = (kk < 8) ? (_Float16)W_agg[(kk + ((lane >> 4) << 3)) * (PD * PD) + c] : (_Float16)0.0f;
    }
}

// ---------------------------------------------------------------------------
// Kernel 1: fused  relu(edges@W_agg + b_agg) -> per-edge matvec -> mask -> sum_j
// One workgroup (4 waves) per (b, i). Each wave owns col-tiles ct = w, w+4, ...
// agg_out[(b*96 + i)*32 + p]
// ---------------------------------------------------------------------------
__global__ __launch_bounds__(128) void msg_agg_kernel(
    const float* __restrict__ nodes,   // [B, N, D]
    const float* __restrict__ edges,   // [B, N*N, F]
    const float* __restrict__ mask,    // [B, N*N, 1]
    const _Float16* __restrict__ bprep,// [64][32][16] f16 B-fragments of W_agg
    const float* __restrict__ b_agg,   // [D*D]
    float* __restrict__ agg_out)       // [B, N, D]
{
    __shared__ float e_lds[PN * PF];   // edge rows for this (b,i): 96 x 16
    __shared__ float smat[PN * PD];    // nodes[b,j,q] * mask[b,i,j]: 96 x 32
    __shared__ float sagg[PD];         // output accumulator

    const int b = blockIdx.x / PN;
    const int i = blockIdx.x - b * PN;
    const int tid = threadIdx.x;

    // --- async stage: edges (384 float4 granules) + nodes (768) -> LDS ---
    {
        const float* esrc = edges + ((size_t)b * (PN * PN) + (size_t)i * PN) * PF;
        const float* nsrc = nodes + (size_t)b * PN * PD;
        #pragma unroll
        for (int k = 0; k < 3; ++k) {                 // 3*128 = 384 granules
            int g = tid + k * 128;
            async_copy_b128(&e_lds[g * 4], esrc + (size_t)g * 4);
        }
        #pragma unroll
        for (int k = 0; k < 6; ++k) {                 // 6*128 = 768 granules
            int g = tid + k * 128;
            async_copy_b128(&smat[g * 4], nsrc + (size_t)g * 4);
        }
    }
    if (tid < PD) sagg[tid] = 0.0f;
    wait_asynccnt0();
    __syncthreads();

    { // fold mask into the state multiplier, in place
        const float* mrow = mask + (size_t)b * (PN * PN) + (size_t)i * PN;
        for (int t = tid; t < PN * PD; t += 128) smat[t] *= mrow[t >> 5];
    }
    __syncthreads();

    const int wave = tid >> 5, lane = tid & 31;
    const int l15 = lane & 15, lh = lane >> 4;

    for (int ct = wave; ct < 64; ct += 4) {          // 64 column tiles of 16
        // B fragment: one contiguous 32B load per lane
        const v16h bf = *(const v16h*)(bprep + (size_t)((ct << 5) + lane) * 16);
        const int   c    = (ct << 4) + l15;          // column of D*D this lane holds
        const float bias = b_agg[c];                 // per-column bias via C operand
        const int   q    = c & 31;                   // q index within D
        const v8f   cf   = {bias, bias, bias, bias, bias, bias, bias, bias};
        float acc = 0.0f;

        #pragma unroll
        for (int jt = 0; jt < 6; ++jt) {             // 6 edge tiles of 16 (j dim)
            // A fragment: lane holds edge row (jt*16 + l15), features 8*lh .. +7
            const float4 f0 = *(const float4*)&e_lds[((jt << 4) + l15) * PF + (lh << 3)];
            const float4 f1 = *(const float4*)&e_lds[((jt << 4) + l15) * PF + (lh << 3) + 4];
            v16h af;
            af[0] = (_Float16)f0.x; af[1] = (_Float16)f0.y;
            af[2] = (_Float16)f0.z; af[3] = (_Float16)f0.w;
            af[4] = (_Float16)f1.x; af[5] = (_Float16)f1.y;
            af[6] = (_Float16)f1.z; af[7] = (_Float16)f1.w;
            #pragma unroll
            for (int u = 8; u < 16; ++u) af[u] = (_Float16)0.0f;  // K=16..31 pad

            v8f df = __builtin_amdgcn_wmma_f32_16x16x32_f16(
                false, af, false, bf, (short)0, cf, false, false);

            // relu, scale by masked state, reduce over (edge, q) in-register
            #pragma unroll
            for (int r = 0; r < 8; ++r) {
                const float v = relu1(df[r]);                  // one v_med3
                const int j = (jt << 4) + r + (lh << 3);       // D-row -> edge j
                acc += v * smat[(j << 5) + q];
            }
        }
        // cross-lane reduction (wave32) -> scalar contribution to agg[p]
        acc += __shfl_xor(acc, 16, 32);
        acc += __shfl_xor(acc,  8, 32);
        acc += __shfl_xor(acc,  4, 32);
        acc += __shfl_xor(acc,  2, 32);
        acc += __shfl_xor(acc,  1, 32);
        if (lane == 0) atomicAdd(&sagg[ct >> 1], acc);         // p = c>>5 = ct>>1
    }
    __syncthreads();
    if (tid < PD) agg_out[(size_t)blockIdx.x * PD + tid] = sagg[tid];
}

// ---------------------------------------------------------------------------
// Kernel 2: two GRU cells via WMMA (K = D = 32 exactly). 1 wave per 16 rows.
// ---------------------------------------------------------------------------
__device__ __forceinline__ float sigm(float x) { return 1.0f / (1.0f + expf(-x)); }

__device__ __forceinline__ v16h loadA_f32(const float* base, int l15, int lh) {
    // A 16x32: lane holds row l15, K = 8*lh + {0..7} and 8*lh + 16 + {0..7}
    const float* p = base + (size_t)l15 * PD + (lh << 3);
    v16h a;
    #pragma unroll
    for (int u = 0; u < 8; ++u) { a[u] = (_Float16)p[u]; a[8 + u] = (_Float16)p[16 + u]; }
    return a;
}

__device__ __forceinline__ v16h loadB_w(const float* w, int gt, int l15, int lh) {
    // gi = x @ w.T : B[k][g] = w[g][k]; lane holds column g = gt*16+l15,
    // K = 8*lh + {0..7} and 8*lh + 16 + {0..7} of row g.
    const float* p = w + (size_t)(gt * 16 + l15) * PD + (lh << 3);
    v16h bv;
    #pragma unroll
    for (int u = 0; u < 8; ++u) { bv[u] = (_Float16)p[u]; bv[8 + u] = (_Float16)p[16 + u]; }
    return bv;
}

__global__ __launch_bounds__(32) void gru_kernel(
    const float* __restrict__ nodes, const float* __restrict__ aggm,
    const float* __restrict__ w_ih,  const float* __restrict__ w_hh,
    const float* __restrict__ b_ih,  const float* __restrict__ b_hh,
    float* __restrict__ out)
{
    __shared__ float    gi[16 * 96];
    __shared__ float    gh[16 * 96];
    __shared__ float    h1[16 * PD];
    __shared__ _Float16 h1h[16 * PD];

    const int lane = threadIdx.x, l15 = lane & 15, lh = lane >> 4;
    const int rowbase = blockIdx.x * 16;   // rows = b*96 + i, 3072 total

    // ---- cell 1: gi1 = nodes @ w_ih.T + b_ih ; gh1 = b_hh (h0 == 0) ----
    v16h ax = loadA_f32(nodes + (size_t)rowbase * PD, l15, lh);
    #pragma unroll
    for (int gt = 0; gt < 6; ++gt) {
        v16h bw = loadB_w(w_ih, gt, l15, lh);
        const int g = gt * 16 + l15;
        const float bi = b_ih[g];
        const v8f c = {bi, bi, bi, bi, bi, bi, bi, bi};
        v8f d = __builtin_amdgcn_wmma_f32_16x16x32_f16(false, ax, false, bw, (short)0, c, false, false);
        #pragma unroll
        for (int r = 0; r < 8; ++r) gi[(r + (lh << 3)) * 96 + g] = d[r];
    }
    __syncthreads();
    { // activations for cell 1: lane covers (row l15, dims 16*lh..+15)
        const int m = l15, dbase = lh << 4;
        #pragma unroll
        for (int dd = 0; dd < 16; ++dd) {
            const int dI = dbase + dd;
            const float r = sigm(gi[m * 96 + dI]      + b_hh[dI]);
            const float z = sigm(gi[m * 96 + 32 + dI] + b_hh[32 + dI]);
            const float n = tanhf(gi[m * 96 + 64 + dI] + r * b_hh[64 + dI]);
            const float h = (1.0f - z) * n;
            h1[m * PD + dI] = h;
            h1h[m * PD + dI] = (_Float16)h;
        }
    }
    __syncthreads();

    // ---- cell 2: gi2 = agg @ w_ih.T + b_ih ; gh2 = h1 @ w_hh.T + b_hh ----
    v16h aagg = loadA_f32(aggm + (size_t)rowbase * PD, l15, lh);
    v16h ah;
    {
        const _Float16* p = h1h + (size_t)l15 * PD + (lh << 3);
        #pragma unroll
        for (int u = 0; u < 8; ++u) { ah[u] = p[u]; ah[8 + u] = p[16 + u]; }
    }
    #pragma unroll
    for (int gt = 0; gt < 6; ++gt) {
        const int g = gt * 16 + l15;
        {
            v16h bw = loadB_w(w_ih, gt, l15, lh);
            const float bi = b_ih[g];
            const v8f c = {bi, bi, bi, bi, bi, bi, bi, bi};
            v8f d = __builtin_amdgcn_wmma_f32_16x16x32_f16(false, aagg, false, bw, (short)0, c, false, false);
            #pragma unroll
            for (int r = 0; r < 8; ++r) gi[(r + (lh << 3)) * 96 + g] = d[r];
        }
        {
            v16h bw = loadB_w(w_hh, gt, l15, lh);
            const float bh = b_hh[g];
            const v8f c = {bh, bh, bh, bh, bh, bh, bh, bh};
            v8f d = __builtin_amdgcn_wmma_f32_16x16x32_f16(false, ah, false, bw, (short)0, c, false, false);
            #pragma unroll
            for (int r = 0; r < 8; ++r) gh[(r + (lh << 3)) * 96 + g] = d[r];
        }
    }
    __syncthreads();
    { // final activations + output
        const int m = l15, dbase = lh << 4;
        #pragma unroll
        for (int dd = 0; dd < 16; ++dd) {
            const int dI = dbase + dd;
            const float r = sigm(gi[m * 96 + dI]      + gh[m * 96 + dI]);
            const float z = sigm(gi[m * 96 + 32 + dI] + gh[m * 96 + 32 + dI]);
            const float n = tanhf(gi[m * 96 + 64 + dI] + r * gh[m * 96 + 64 + dI]);
            out[(size_t)(rowbase + m) * PD + dI] = (1.0f - z) * n + z * h1[m * PD + dI];
        }
    }
}

// ---------------------------------------------------------------------------
extern "C" void kernel_launch(void* const* d_in, const int* in_sizes, int n_in,
                              void* d_out, int out_size, void* d_ws, size_t ws_size,
                              hipStream_t stream) {
    (void)in_sizes; (void)n_in; (void)out_size; (void)ws_size;
    const float* nodes = (const float*)d_in[0];
    const float* edges = (const float*)d_in[1];
    const float* mask  = (const float*)d_in[2];
    const float* W_agg = (const float*)d_in[3];
    const float* b_agg = (const float*)d_in[4];
    const float* w_ih  = (const float*)d_in[5];
    const float* w_hh  = (const float*)d_in[6];
    const float* b_ih  = (const float*)d_in[7];
    const float* b_hh  = (const float*)d_in[8];
    float* out = (float*)d_out;

    // workspace: [0, 64KB) = W_agg f16 B-fragments; [64KB, ...) = agg_m f32
    _Float16* bprep = (_Float16*)d_ws;
    float*    aggm  = (float*)((char*)d_ws + 64 * 1024);

    prep_wagg_kernel<<<8, 256, 0, stream>>>(W_agg, bprep);
    msg_agg_kernel<<<PB * PN, 128, 0, stream>>>(nodes, edges, mask, bprep, b_agg, aggm);
    gru_kernel<<<(PB * PN) / 16, 32, 0, stream>>>(nodes, aggm, w_ih, w_hh, b_ih, b_hh, out);
}
